// AttentionBlock_89627377533209
// MI455X (gfx1250) — matrix operand order repaired and checked by reading, not compiled, not run on hardware
//
#include <hip/hip_runtime.h>
#include <hip/hip_bf16.h>

#define DD 768
#define NN 2048
#define BB 8

typedef __attribute__((ext_vector_type(16))) __bf16 v16bf;
typedef __attribute__((ext_vector_type(8)))  __bf16 v8bf;
typedef __attribute__((ext_vector_type(8)))  float  v8f;

union ABFrag { v16bf v; v8bf h[2]; };

// ---------------------------------------------------------------- convert
__global__ void cvt_f32_bf16(const float* __restrict__ src,
                             __bf16* __restrict__ dst, int n) {
  int i = blockIdx.x * blockDim.x + threadIdx.x;
  int stride = gridDim.x * blockDim.x;
  for (; i < n; i += stride) dst[i] = (__bf16)src[i];
}

// ---------------------------------------------------------------- QKV GEMM
// One wave computes a 16x64 tile (4 column sub-tiles sharing one A fragment)
// of Q, K or V (blockIdx.y selects which). Q,K stored [b][n][d] bf16;
// V stored transposed Vt[b][e][seq] so P@V B-fragments are seq-contiguous.
__global__ __launch_bounds__(256) void qkv_gemm(
    const __bf16* __restrict__ xh,
    const __bf16* __restrict__ Wqh, const __bf16* __restrict__ Wkh,
    const __bf16* __restrict__ Wvh,
    const float* __restrict__ bq, const float* __restrict__ bk,
    const float* __restrict__ bv,
    __bf16* __restrict__ Qh, __bf16* __restrict__ Kh,
    __bf16* __restrict__ Vt) {
  const int lane  = threadIdx.x & 31;
  const int wave  = threadIdx.x >> 5;
  const int which = blockIdx.y;                 // 0=Q 1=K 2=V
  const int NTQ   = DD / 64;                    // 12 quad col tiles
  const int t     = blockIdx.x * 8 + wave;      // wave-tile id
  const int mt    = t / NTQ;                    // 0..(B*N/16-1)
  const int ntq   = t % NTQ;

  const __bf16* W    = (which == 0) ? Wqh : (which == 1) ? Wkh : Wvh;
  const float*  bias = (which == 0) ? bq  : (which == 1) ? bk  : bv;

  const int r0     = mt * 16;                   // flattened over B*N
  const int rowA   = r0 + (lane & 15);
  const int kbaseA = (lane >> 4) * 16;
  const int kbaseB = (lane >> 4) * 16;

  v8f zero = {};
  v8f c[4];
#pragma unroll
  for (int q = 0; q < 4; ++q) c[q] = zero;

  const __bf16* Arow = xh + (size_t)rowA * DD;
  const __bf16* Brow[4];
#pragma unroll
  for (int q = 0; q < 4; ++q)
    Brow[q] = W + (size_t)((ntq * 4 + q) * 16 + (lane & 15)) * DD;

#pragma unroll
  for (int kt = 0; kt < DD / 32; ++kt) {
    ABFrag a;
    a.h[0] = *(const v8bf*)(Arow + kt * 32 + kbaseA);
    a.h[1] = *(const v8bf*)(Arow + kt * 32 + kbaseA + 8);
#pragma unroll
    for (int q = 0; q < 4; ++q) {
      ABFrag b;
      b.h[0] = *(const v8bf*)(Brow[q] + kt * 32 + kbaseB);
      b.h[1] = *(const v8bf*)(Brow[q] + kt * 32 + kbaseB + 8);
      c[q] = __builtin_amdgcn_wmma_f32_16x16x32_bf16(false, a.v, false, b.v,
                                                     (short)0, c[q], false, false);
    }
  }

  const int mbase = (lane >> 4) * 8;
#pragma unroll
  for (int q = 0; q < 4; ++q) {
    const int   ncol = (ntq * 4 + q) * 16 + (lane & 15);
    const float bval = bias[ncol];
    if (which == 2) {
      // transposed store: Vt[b][e][seq]; the 8 C values are seq-contiguous.
      const int bidx = r0 / NN;
      const int seq0 = (r0 % NN) + mbase;
      v8bf pk;
#pragma unroll
      for (int i = 0; i < 8; ++i) pk[i] = (__bf16)(c[q][i] + bval);
      *(v8bf*)(Vt + (size_t)bidx * DD * NN + (size_t)ncol * NN + seq0) = pk;
    } else {
      __bf16* O = (which == 0) ? Qh : Kh;
#pragma unroll
      for (int i = 0; i < 8; ++i)
        O[(size_t)(r0 + mbase + i) * DD + ncol] = (__bf16)(c[q][i] + bval);
    }
  }
}

// ---------------------------------------------------------------- attention
// One block (8 waves) per 16-row query tile. LDS: S f32 (128KB) + P bf16 (64KB).
__global__ __launch_bounds__(256) void attn(
    const __bf16* __restrict__ Qh, const __bf16* __restrict__ Kh,
    const __bf16* __restrict__ Vt, const float* __restrict__ x,
    const float* __restrict__ gptr, float* __restrict__ out) {
  extern __shared__ char smem[];
  float*  Sf = (float*)smem;                              // 16 x 2048 f32
  __bf16* Ph = (__bf16*)(smem + 16 * NN * sizeof(float)); // 16 x 2048 bf16

  const int lane = threadIdx.x & 31;
  const int wave = threadIdx.x >> 5;
  const int bidx = blockIdx.x >> 7;   // / (N/16)
  const int qt   = blockIdx.x & 127;
  const int r0   = qt * 16;

  const __bf16* Qb = Qh + (size_t)bidx * NN * DD;
  const __bf16* Kb = Kh + (size_t)bidx * NN * DD;
  const __bf16* Vb = Vt + (size_t)bidx * DD * NN;

  const int rowA   = r0 + (lane & 15);
  const int kbaseA = (lane >> 4) * 16;
  const int kbaseB = (lane >> 4) * 16;
  const int ncol   = lane & 15;
  const int mbase  = (lane >> 4) * 8;

  const __bf16* Arow = Qb + (size_t)rowA * DD;

  // ---- Phase 1: S = Q @ K^T  (16 x 2048, f32 in LDS)
  // Each wave handles 4 adjacent key tiles per pass sharing one A fragment.
  for (int it = 0; it < 4; ++it) {
    const int j0 = it * 32 + wave * 4;            // first of 4 key tiles
    v8f zero = {};
    v8f c[4];
#pragma unroll
    for (int q = 0; q < 4; ++q) c[q] = zero;
    const __bf16* Brow[4];
#pragma unroll
    for (int q = 0; q < 4; ++q)
      Brow[q] = Kb + (size_t)((j0 + q) * 16 + ncol) * DD;
    // keep next pass's K stream warm in L0/L2
    if (it < 3)
      __builtin_prefetch(Kb + (size_t)((j0 + 32) * 16 + ncol) * DD, 0, 3);
#pragma unroll
    for (int kt = 0; kt < DD / 32; ++kt) {
      ABFrag a;
      a.h[0] = *(const v8bf*)(Arow + kt * 32 + kbaseA);
      a.h[1] = *(const v8bf*)(Arow + kt * 32 + kbaseA + 8);
#pragma unroll
      for (int q = 0; q < 4; ++q) {
        ABFrag b;
        b.h[0] = *(const v8bf*)(Brow[q] + kt * 32 + kbaseB);
        b.h[1] = *(const v8bf*)(Brow[q] + kt * 32 + kbaseB + 8);
        c[q] = __builtin_amdgcn_wmma_f32_16x16x32_bf16(
            false, a.v, false, b.v, (short)0, c[q], false, false);
      }
    }
#pragma unroll
    for (int q = 0; q < 4; ++q)
#pragma unroll
      for (int i = 0; i < 8; ++i)
        Sf[(mbase + i) * NN + (j0 + q) * 16 + ncol] = c[q][i];
  }
  __syncthreads();

  // ---- Phase 2: row softmax (each wave owns 2 rows)
  for (int rr = 0; rr < 2; ++rr) {
    const int row = wave * 2 + rr;
    float mx = -3.4e38f;
    for (int k = lane; k < NN; k += 32) mx = fmaxf(mx, Sf[row * NN + k]);
#pragma unroll
    for (int off = 16; off; off >>= 1) mx = fmaxf(mx, __shfl_xor(mx, off, 32));
    float sum = 0.f;
    for (int k = lane; k < NN; k += 32) {
      float e = __expf(Sf[row * NN + k] - mx);
      Sf[row * NN + k] = e;
      sum += e;
    }
#pragma unroll
    for (int off = 16; off; off >>= 1) sum += __shfl_xor(sum, off, 32);
    const float inv = 1.0f / sum;
    for (int k = lane; k < NN; k += 32)
      Ph[row * NN + k] = (__bf16)(Sf[row * NN + k] * inv);
  }
  __syncthreads();

  // ---- Phase 3: O = P @ V ; each wave owns 6 of the 48 column tiles,
  // sharing one LDS A-fragment across 6 WMMAs per K-step.
  v8f zero = {};
  v8f o[6];
#pragma unroll
  for (int t = 0; t < 6; ++t) o[t] = zero;
  const int ct0 = wave * 6;
  for (int kk = 0; kk < NN / 32; ++kk) {
    ABFrag a;
    const int kstart = kk * 32 + kbaseA;
    a.h[0] = *(const v8bf*)(Ph + (lane & 15) * NN + kstart);
    a.h[1] = *(const v8bf*)(Ph + (lane & 15) * NN + kstart + 8);
#pragma unroll
    for (int t = 0; t < 6; ++t) {
      ABFrag b;
      const int e = (ct0 + t) * 16 + ncol;
      b.h[0] = *(const v8bf*)(Vb + (size_t)e * NN + kk * 32 + kbaseB);
      b.h[1] = *(const v8bf*)(Vb + (size_t)e * NN + kk * 32 + kbaseB + 8);
      o[t] = __builtin_amdgcn_wmma_f32_16x16x32_bf16(false, a.v, false, b.v,
                                                     (short)0, o[t], false, false);
    }
  }

  // ---- Epilogue: out = gamma * O + x
  const float gamma = gptr[0];
#pragma unroll
  for (int t = 0; t < 6; ++t) {
    const int e = (ct0 + t) * 16 + ncol;
#pragma unroll
    for (int i = 0; i < 8; ++i) {
      const size_t g = ((size_t)bidx * NN + r0 + mbase + i) * DD + e;
      out[g] = gamma * o[t][i] + x[g];
    }
  }
}

// ---------------------------------------------------------------- launch
extern "C" void kernel_launch(void* const* d_in, const int* in_sizes, int n_in,
                              void* d_out, int out_size, void* d_ws,
                              size_t ws_size, hipStream_t stream) {
  const float* x  = (const float*)d_in[0];
  const float* Wq = (const float*)d_in[1];
  const float* bq = (const float*)d_in[2];
  const float* Wk = (const float*)d_in[3];
  const float* bk = (const float*)d_in[4];
  const float* Wv = (const float*)d_in[5];
  const float* bv = (const float*)d_in[6];
  const float* gm = (const float*)d_in[7];
  float* out = (float*)d_out;

  char* ws = (char*)d_ws;
  size_t off = 0;
  __bf16* xh  = (__bf16*)(ws + off); off += (size_t)BB * NN * DD * 2;
  __bf16* Wqh = (__bf16*)(ws + off); off += (size_t)DD * DD * 2;
  __bf16* Wkh = (__bf16*)(ws + off); off += (size_t)DD * DD * 2;
  __bf16* Wvh = (__bf16*)(ws + off); off += (size_t)DD * DD * 2;
  __bf16* Qh  = (__bf16*)(ws + off); off += (size_t)BB * NN * DD * 2;
  __bf16* Kh  = (__bf16*)(ws + off); off += (size_t)BB * NN * DD * 2;
  __bf16* Vt  = (__bf16*)(ws + off); off += (size_t)BB * NN * DD * 2;

  cvt_f32_bf16<<<1024, 256, 0, stream>>>(x, xh, BB * NN * DD);
  cvt_f32_bf16<<<256, 256, 0, stream>>>(Wq, Wqh, DD * DD);
  cvt_f32_bf16<<<256, 256, 0, stream>>>(Wk, Wkh, DD * DD);
  cvt_f32_bf16<<<256, 256, 0, stream>>>(Wv, Wvh, DD * DD);

  // (B*N/16) row tiles * (D/64) quad col tiles, 8 waves/block, y = {Q,K,V}
  const int wtiles = (BB * NN / 16) * (DD / 64);   // 12288
  qkv_gemm<<<dim3(wtiles / 8, 3), 256, 0, stream>>>(
      xh, Wqh, Wkh, Wvh, bq, bk, bv, Qh, Kh, Vt);

  const size_t smem = 16 * NN * sizeof(float) + 16 * NN * sizeof(__bf16);
  attn<<<dim3(BB * NN / 16), 256, smem, stream>>>(Qh, Kh, Vt, x, gm, out);
}
